// MultiScaleCodebook_65395172049441
// MI455X (gfx1250) — matrix operand order repaired and checked by reading, not compiled, not run on hardware
//
#include <hip/hip_runtime.h>
#include <hip/hip_bf16.h>

// ---------------- CDNA5 WMMA types ----------------
typedef __attribute__((ext_vector_type(16))) __bf16        v16bf;
typedef __attribute__((ext_vector_type(8)))  float         v8f;
typedef __attribute__((ext_vector_type(8)))  unsigned int  v8u;

union BFrag { v8u u; v16bf b; };

#define N_ROWS   32768
#define D_K      512
#define N_CODE   1024
#define N_SUB1   512
#define N_SUB2   256

__device__ __forceinline__ unsigned int f2bf(float f) {
    unsigned int u = __float_as_uint(f);
    u += 0x7fffu + ((u >> 16) & 1u);   // round-to-nearest-even
    return u >> 16;
}

// ---------------- init: bests=+inf keys, hist=0, loss=0, out tail zeros ----------------
__global__ void k_init(unsigned long long* __restrict__ best, int nbest,
                       unsigned int* __restrict__ hist,
                       float* __restrict__ lossacc,
                       float* __restrict__ out_tail, int ntail) {
    int i = blockIdx.x * blockDim.x + threadIdx.x;
    if (i < nbest) best[i] = ~0ull;
    if (i < N_CODE) hist[i] = 0u;
    if (i < 2) lossacc[i] = 0.0f;
    if (i < ntail) out_tail[i] = 0.0f;
}

// ---------------- x -> bf16, 4 elements / thread ----------------
__global__ void k_xbf(const float4* __restrict__ x4, uint2* __restrict__ o2, int n4) {
    int i = blockIdx.x * blockDim.x + threadIdx.x;
    if (i < n4) {
        float4 v = x4[i];
        uint2 o;
        o.x = f2bf(v.x) | (f2bf(v.y) << 16);
        o.y = f2bf(v.z) | (f2bf(v.w) << 16);
        o2[i] = o;
    }
}

// ---------------- fp32 codebook -> transposed bf16 [D_K][nrows] + row norms ----------------
__global__ __launch_bounds__(256) void k_prep_cb(const float* __restrict__ src,
                                                 unsigned short* __restrict__ dstT,
                                                 float* __restrict__ norms, int nrows) {
    int r = blockIdx.x;
    __shared__ float red[256];
    float s = 0.0f;
    for (int d = threadIdx.x; d < D_K; d += 256) {
        float v = src[(size_t)r * D_K + d];
        s += v * v;
        dstT[(size_t)d * nrows + r] = (unsigned short)f2bf(v);
    }
    red[threadIdx.x] = s; __syncthreads();
    for (int o = 128; o > 0; o >>= 1) {
        if (threadIdx.x < o) red[threadIdx.x] += red[threadIdx.x + o];
        __syncthreads();
    }
    if (threadIdx.x == 0) norms[r] = red[0];
}

// ---------------- sub_c = W @ c + b (fp32) ----------------
__global__ __launch_bounds__(256) void k_subc(const float* __restrict__ W,
                                              const float* __restrict__ bias,
                                              const float* __restrict__ c,
                                              float* __restrict__ out) {
    int d = blockIdx.x * 256 + threadIdx.x;   // 0..511, grid.x = 2
    int o = blockIdx.y;                        // output row
    float acc = 0.0f;
    for (int k = 0; k < N_CODE; ++k)
        acc = fmaf(W[(size_t)o * N_CODE + k], c[(size_t)k * D_K + d], acc);
    out[(size_t)o * D_K + d] = acc + bias[o];
}

// ---------------- WMMA bf16 GEMM + argmin(||c||^2 - 2 x.c) ----------------
// block = 128 (4 waves). Each wave: 32 rows (2 row-tiles) x 64 cols (4 col-tiles).
// grid = (N_ROWS/32, n_code/256)
__global__ __launch_bounds__(128) void k_gemm_argmin(
    const unsigned short* __restrict__ xbf,   // [N_ROWS][D_K]
    const unsigned short* __restrict__ cbT,   // [D_K][n_code]
    const float* __restrict__ norms,          // [n_code]
    unsigned long long* __restrict__ best,    // [N_ROWS]
    int n_code) {
    const int lane    = threadIdx.x & 31;
    const int wave    = threadIdx.x >> 5;
    const int l16     = lane & 15;
    const int khalf   = lane >> 4;
    const int rowbase = blockIdx.x * 32;
    const int colbase = blockIdx.y * 256 + wave * 64;

    v8f acc0[4], acc1[4];
    #pragma unroll
    for (int t = 0; t < 4; ++t) {
        acc0[t] = (v8f){0.f,0.f,0.f,0.f,0.f,0.f,0.f,0.f};
        acc1[t] = (v8f){0.f,0.f,0.f,0.f,0.f,0.f,0.f,0.f};
    }

    const unsigned int* arow0 =
        (const unsigned int*)(xbf + (size_t)(rowbase + l16) * D_K);
    const unsigned int* arow1 = arow0 + 16 * (D_K / 2);   // +16 rows

    #pragma unroll 2
    for (int kb = 0; kb < D_K; kb += 32) {
        // ---- load phase: A (2 row-tiles) + B (4 col-tiles), no WMMA deps yet ----
        BFrag a0, a1, bfr[4];
        const unsigned int* ap0 = arow0 + (kb >> 1);
        const unsigned int* ap1 = arow1 + (kb >> 1);
        #pragma unroll
        for (int v = 0; v < 4; ++v) {
            a0.u[v] = ap0[v + 4 * khalf];
            a1.u[v] = ap1[v + 4 * khalf];
        }
        #pragma unroll
        for (int v = 4; v < 8; ++v) {
            a0.u[v] = ap0[v + 4 + 4 * khalf];
            a1.u[v] = ap1[v + 4 + 4 * khalf];
        }
        const unsigned short* bcol =
            cbT + (size_t)kb * n_code + colbase + l16;
        #pragma unroll
        for (int t = 0; t < 4; ++t) {
            const unsigned short* bp = bcol + t * 16;
            #pragma unroll
            for (int v = 0; v < 8; ++v) {
                int k = 2 * v + 8 * khalf + (v >= 4 ? 8 : 0);
                bfr[t].u[v] = (unsigned int)bp[(size_t)k * n_code] |
                              ((unsigned int)bp[(size_t)(k + 1) * n_code] << 16);
            }
        }
        // ---- compute phase: 8 WMMAs, B reused twice, A reused 4x ----
        #pragma unroll
        for (int t = 0; t < 4; ++t) {
            acc0[t] = __builtin_amdgcn_wmma_f32_16x16x32_bf16(
                false, a0.b, false, bfr[t].b, (short)0, acc0[t], false, false);
            acc1[t] = __builtin_amdgcn_wmma_f32_16x16x32_bf16(
                false, a1.b, false, bfr[t].b, (short)0, acc1[t], false, false);
        }
    }

    // Epilogue: D layout — VGPR j holds row (j + 8*khalf), lane l16 = column.
    const int ncol0 = colbase + l16;
    #pragma unroll
    for (int rt = 0; rt < 2; ++rt) {
        v8f* acc = rt ? acc1 : acc0;
        #pragma unroll
        for (int j = 0; j < 8; ++j) {
            float bestv = norms[ncol0] - 2.0f * acc[0][j];
            int   besti = ncol0;
            #pragma unroll
            for (int t = 1; t < 4; ++t) {
                int n = ncol0 + t * 16;
                float cand = norms[n] - 2.0f * acc[t][j];
                if (cand < bestv) { bestv = cand; besti = n; }
            }
            // shuffle argmin across 16 columns (stays within lane half)
            #pragma unroll
            for (int m = 1; m < 16; m <<= 1) {
                float ov = __shfl_xor(bestv, m, 32);
                int   oi = __shfl_xor(besti, m, 32);
                if (ov < bestv || (ov == bestv && oi < besti)) { bestv = ov; besti = oi; }
            }
            if (l16 == 0) {
                unsigned int fb = __float_as_uint(bestv);
                fb = (fb & 0x80000000u) ? ~fb : (fb ^ 0x80000000u);  // order-preserving
                unsigned long long key =
                    ((unsigned long long)fb << 32) | (unsigned int)besti;
                atomicMin(best + rowbase + rt * 16 + j + 8 * khalf, key);
            }
        }
    }
}

// ---------------- gather outputs + MSE sums + usage histogram ----------------
__global__ __launch_bounds__(256) void k_finalize(
    const float* __restrict__ x, const float* __restrict__ c,
    const float* __restrict__ s1, const float* __restrict__ s2,
    const unsigned long long* __restrict__ bm,
    const unsigned long long* __restrict__ b1,
    const unsigned long long* __restrict__ b2,
    float* __restrict__ main_out, float* __restrict__ sub_out,
    unsigned int* __restrict__ hist, float* __restrict__ lossacc) {
    int row = blockIdx.x;
    int im = (int)(bm[row] & 0xffffffffull) & (N_CODE - 1);
    int i1 = (int)(b1[row] & 0xffffffffull) & (N_SUB1 - 1);
    int i2 = (int)(b2[row] & 0xffffffffull) & (N_SUB2 - 1);
    const float* xr = x  + (size_t)row * D_K;
    const float* cm = c  + (size_t)im  * D_K;
    const float* c1 = s1 + (size_t)i1  * D_K;
    const float* c2 = s2 + (size_t)i2  * D_K;
    float lm = 0.0f, ls = 0.0f;
    for (int d = threadIdx.x; d < D_K; d += 256) {
        float xv = xr[d];
        float mo = cm[d];
        float so = 0.5f * (c1[d] + c2[d]);
        main_out[(size_t)row * D_K + d] = mo;
        sub_out [(size_t)row * D_K + d] = so;
        float dm = mo - xv, ds = so - xv;
        lm += dm * dm; ls += ds * ds;
    }
    __shared__ float r0[256], r1[256];
    r0[threadIdx.x] = lm; r1[threadIdx.x] = ls; __syncthreads();
    for (int o = 128; o > 0; o >>= 1) {
        if (threadIdx.x < o) {
            r0[threadIdx.x] += r0[threadIdx.x + o];
            r1[threadIdx.x] += r1[threadIdx.x + o];
        }
        __syncthreads();
    }
    if (threadIdx.x == 0) {
        atomicAdd(lossacc,     r0[0]);
        atomicAdd(lossacc + 1, r1[0]);
        hist[im] = 1u;
    }
}

// ---------------- scalars: total_loss, unique_rows ----------------
__global__ __launch_bounds__(256) void k_scalars(const unsigned int* __restrict__ hist,
                                                 const float* __restrict__ lossacc,
                                                 const int* __restrict__ training,
                                                 float* __restrict__ loss_out,
                                                 float* __restrict__ unique_out) {
    __shared__ unsigned int red[256];
    unsigned int cnt = 0;
    for (int i = threadIdx.x; i < N_CODE; i += 256) cnt += (hist[i] != 0u);
    red[threadIdx.x] = cnt; __syncthreads();
    for (int o = 128; o > 0; o >>= 1) {
        if (threadIdx.x < o) red[threadIdx.x] += red[threadIdx.x + o];
        __syncthreads();
    }
    if (threadIdx.x == 0) {
        float inv = 1.0f / ((float)N_ROWS * (float)D_K);
        float tl  = 1.25f * (lossacc[0] + lossacc[1]) * inv;  // (0.25+1.0)*(e_m + e_s)
        *loss_out   = (*training != 0) ? tl : 0.0f;
        *unique_out = (float)red[0];
    }
}

extern "C" void kernel_launch(void* const* d_in, const int* in_sizes, int n_in,
                              void* d_out, int out_size, void* d_ws, size_t ws_size,
                              hipStream_t stream) {
    const float* x  = (const float*)d_in[0];   // [1, 32768, 512]
    const float* c  = (const float*)d_in[1];   // [1024, 512]
    const float* W1 = (const float*)d_in[2];   // [512, 1024]
    const float* b1 = (const float*)d_in[3];   // [512]
    const float* W2 = (const float*)d_in[4];   // [256, 1024]
    const float* b2 = (const float*)d_in[5];   // [256]
    const int* training = (const int*)d_in[6];

    float* out = (float*)d_out;
    float* main_out = out;                                   // 16777216
    float* sub_out  = out + (size_t)N_ROWS * D_K;            // 16777216
    float* loss_out = out + (size_t)2 * N_ROWS * D_K;        // 1
    float* out_tail = loss_out;                               // loss + ms_k_i + unique
    const int ntail = 1 + 16 * D_K + 1;                       // 8194
    float* unique_out = loss_out + 1 + 16 * D_K;

    // ---- workspace layout ----
    char* w = (char*)d_ws;
    size_t off = 0;
    auto take = [&](size_t bytes) { char* p = w + off; off = (off + bytes + 255) & ~(size_t)255; return p; };
    unsigned short* xbf   = (unsigned short*)take((size_t)N_ROWS * D_K * 2);   // 32 MB
    unsigned short* cT    = (unsigned short*)take((size_t)D_K * N_CODE * 2);
    unsigned short* s1T   = (unsigned short*)take((size_t)D_K * N_SUB1 * 2);
    unsigned short* s2T   = (unsigned short*)take((size_t)D_K * N_SUB2 * 2);
    float* s1f            = (float*)take((size_t)N_SUB1 * D_K * 4);
    float* s2f            = (float*)take((size_t)N_SUB2 * D_K * 4);
    float* norm_c         = (float*)take(N_CODE * 4);
    float* norm_1         = (float*)take(N_SUB1 * 4);
    float* norm_2         = (float*)take(N_SUB2 * 4);
    unsigned long long* best = (unsigned long long*)take((size_t)3 * N_ROWS * 8);
    unsigned long long* best_m  = best;
    unsigned long long* best_s1 = best + N_ROWS;
    unsigned long long* best_s2 = best + 2 * N_ROWS;
    unsigned int* hist    = (unsigned int*)take(N_CODE * 4);
    float* lossacc        = (float*)take(2 * 4);

    // 1) init
    {
        int nbest = 3 * N_ROWS;
        k_init<<<(nbest + 255) / 256, 256, 0, stream>>>(best, nbest, hist, lossacc,
                                                        out_tail, ntail);
    }
    // 2) x -> bf16 (vectorized)
    {
        int n4 = N_ROWS * D_K / 4;
        k_xbf<<<(n4 + 255) / 256, 256, 0, stream>>>((const float4*)x, (uint2*)xbf, n4);
    }
    // 3) sub codebooks (fp32)
    k_subc<<<dim3(2, N_SUB1), 256, 0, stream>>>(W1, b1, c, s1f);
    k_subc<<<dim3(2, N_SUB2), 256, 0, stream>>>(W2, b2, c, s2f);
    // 4) transpose+bf16+norms for all three codebooks
    k_prep_cb<<<N_CODE, 256, 0, stream>>>(c,   cT,  norm_c, N_CODE);
    k_prep_cb<<<N_SUB1, 256, 0, stream>>>(s1f, s1T, norm_1, N_SUB1);
    k_prep_cb<<<N_SUB2, 256, 0, stream>>>(s2f, s2T, norm_2, N_SUB2);
    // 5) WMMA GEMM + argmin for each codebook
    k_gemm_argmin<<<dim3(N_ROWS / 32, N_CODE / 256), 128, 0, stream>>>(
        xbf, cT, norm_c, best_m, N_CODE);
    k_gemm_argmin<<<dim3(N_ROWS / 32, N_SUB1 / 256), 128, 0, stream>>>(
        xbf, s1T, norm_1, best_s1, N_SUB1);
    k_gemm_argmin<<<dim3(N_ROWS / 32, N_SUB2 / 256), 128, 0, stream>>>(
        xbf, s2T, norm_2, best_s2, N_SUB2);
    // 6) gather outputs, losses, histogram
    k_finalize<<<N_ROWS, 256, 0, stream>>>(x, c, s1f, s2f, best_m, best_s1, best_s2,
                                           main_out, sub_out, hist, lossacc);
    // 7) scalars
    k_scalars<<<1, 256, 0, stream>>>(hist, lossacc, training, loss_out, unique_out);
}
